// GNNEncoder_81372450390200
// MI455X (gfx1250) — matrix operand order repaired and checked by reading, not compiled, not run on hardware
//
#include <hip/hip_runtime.h>

typedef __attribute__((ext_vector_type(16))) _Float16 v16h;
typedef __attribute__((ext_vector_type(8)))  _Float16 v8h;
typedef __attribute__((ext_vector_type(8)))  float    v8f;
typedef __attribute__((ext_vector_type(4)))  float    v4f;

// ---------------- elementwise helpers ----------------

__global__ void k_zero(float* __restrict__ p, int n) {
  int i = blockIdx.x * blockDim.x + threadIdx.x;
  if (i < n) p[i] = 0.0f;
}

__global__ void k_relu(float* __restrict__ p, int n) {
  int i = blockIdx.x * blockDim.x + threadIdx.x;
  if (i < n) p[i] = fmaxf(p[i], 0.0f);
}

__global__ void k_cvt_f16(const float* __restrict__ in, _Float16* __restrict__ out, int n) {
  int i = blockIdx.x * blockDim.x + threadIdx.x;
  if (i < n) out[i] = (_Float16)in[i];
}

// Pack W [128x128] row-major (W[k*128+n]) into WMMA B-fragment order.
// Flat layout: Wp[((ct*4+kt)*32 + lane)*16 + j]
//   lane holds column N = ct*16 + lane%16
//   half j (0..15) holds K = kt*32 + (lane/16)*8 + (j%8) + (j/8)*16   (ISA 05_wmma B layout)
__global__ void k_pack_w(const float* __restrict__ W, _Float16* __restrict__ Wp) {
  int i = blockIdx.x * blockDim.x + threadIdx.x;   // 16384 threads total
  if (i >= 8 * 4 * 32 * 16) return;
  int j    = i & 15;
  int lane = (i >> 4) & 31;
  int kt   = (i >> 9) & 3;
  int ct   = (i >> 11) & 7;
  int n = ct * 16 + (lane & 15);
  int k = kt * 32 + (lane >> 4) * 8 + (j & 7) + (j >> 3) * 16;
  Wp[i] = (_Float16)W[k * 128 + n];
}

// ---------------- WMMA GEMM: H[N,128] = A16[N,128] @ W(packed) + bias ----------------
// One wave per 16x16 output tile; block = 8 waves = all 8 column tiles of one row tile.
__global__ void k_gemm_wmma(const _Float16* __restrict__ A16,
                            const _Float16* __restrict__ Wp,
                            const float* __restrict__ bias,
                            float* __restrict__ H, int N) {
  int wave = threadIdx.x >> 5;      // colTile 0..7
  int lane = threadIdx.x & 31;
  int rowTile = blockIdx.x;
  int colTile = wave;
  int ln = lane & 15;               // N within tile / M row for A
  int lh = lane >> 4;               // half select

  int base = rowTile * 16;
  int r  = base + ln;
  int rl = (r < N) ? r : (N - 1);   // clamp loads; stores guarded below

  v8f acc = {};
#pragma unroll
  for (int kt = 0; kt < 4; ++kt) {
    // A fragment: two contiguous 16B chunks per lane (K = kt*32 + 8*lh + {0..7}, +16)
    const _Float16* ap = A16 + (size_t)rl * 128 + kt * 32 + lh * 8;
    v8h alo = *(const v8h*)(ap);
    v8h ahi = *(const v8h*)(ap + 16);
    v16h a;
#pragma unroll
    for (int j = 0; j < 8; ++j) { a[j] = alo[j]; a[8 + j] = ahi[j]; }
    // B fragment: pre-packed contiguous 32B per lane
    v16h b = *(const v16h*)(Wp + ((size_t)(colTile * 4 + kt) * 32 + lane) * 16);
    acc = __builtin_amdgcn_wmma_f32_16x16x32_f16(
        /*neg_a=*/false, a, /*neg_b=*/false, b,
        /*c_mod=*/(short)0, acc, /*reuse_a=*/false, /*reuse_b=*/false);
  }

  float bv = bias[colTile * 16 + ln];
  float* __restrict__ Hc = H + (size_t)(base + 8 * lh) * 128 + colTile * 16 + ln;
  if (base + 16 <= N) {
    // Wave-uniform fast path: one scalar branch, 8 unconditional (clause-able) stores.
#pragma unroll
    for (int v = 0; v < 8; ++v) Hc[(size_t)v * 128] = acc[v] + bv;
  } else {
    // Ragged final tile (not hit when N % 16 == 0).
#pragma unroll
    for (int v = 0; v < 8; ++v) {
      int m = base + v + 8 * lh;
      if (m < N) H[(size_t)m * 128 + colTile * 16 + ln] = acc[v] + bv;
    }
  }
}

// ---------------- L2 row normalization (wave32 per row, 4 floats/lane) ----------------
__global__ void k_rownorm(const float* __restrict__ H, float* __restrict__ HN, int N) {
  int wave = threadIdx.x >> 5, lane = threadIdx.x & 31;
  int row = blockIdx.x * 8 + wave;
  if (row >= N) return;
  v4f v = *(const v4f*)(H + (size_t)row * 128 + lane * 4);
  float ss = v.x * v.x + v.y * v.y + v.z * v.z + v.w * v.w;
#pragma unroll
  for (int off = 16; off > 0; off >>= 1) ss += __shfl_xor(ss, off, 32);
  float s = rsqrtf(ss + 1e-12f);
  v4f o = { v.x * s, v.y * s, v.z * s, v.w * s };
  *(v4f*)(HN + (size_t)row * 128 + lane * 4) = o;
}

// ---------------- edge score: p=exp(beta*cos), accumulate softmax denominator -------
// Max-shift elided: |score| <= |beta| so exp cannot overflow; self-loop => denom > 0.
__global__ void k_edge_score(const float* __restrict__ HN,
                             const int* __restrict__ srcI, const int* __restrict__ dstI,
                             const float* __restrict__ beta,
                             float* __restrict__ sc, float* __restrict__ den, int E) {
  int wave = threadIdx.x >> 5, lane = threadIdx.x & 31;
  int e = blockIdx.x * 8 + wave;
  if (e >= E) return;
  int s = srcI[e], d = dstI[e];
  v4f a = *(const v4f*)(HN + (size_t)s * 128 + lane * 4);
  v4f b = *(const v4f*)(HN + (size_t)d * 128 + lane * 4);
  float dot = a.x * b.x + a.y * b.y + a.z * b.z + a.w * b.w;
#pragma unroll
  for (int off = 16; off > 0; off >>= 1) dot += __shfl_xor(dot, off, 32);
  if (lane == 0) {
    float p = __expf(beta[0] * dot);
    sc[e] = p;
    atomicAdd(&den[d], p);
  }
}

// ---------------- weighted scatter: out[dst] += (p/den[dst]) * h[src] ----------------
__global__ void k_edge_agg(const float* __restrict__ H, const float* __restrict__ sc,
                           const float* __restrict__ den,
                           const int* __restrict__ srcI, const int* __restrict__ dstI,
                           float* __restrict__ out, int E) {
  int wave = threadIdx.x >> 5, lane = threadIdx.x & 31;
  int e = blockIdx.x * 8 + wave;
  if (e >= E) return;
  int s = srcI[e], d = dstI[e];
  float alpha = sc[e] / den[d];
  v4f hv = *(const v4f*)(H + (size_t)s * 128 + lane * 4);
  float* o = out + (size_t)d * 128 + lane * 4;
  atomicAdd(o + 0, alpha * hv.x);
  atomicAdd(o + 1, alpha * hv.y);
  atomicAdd(o + 2, alpha * hv.z);
  atomicAdd(o + 3, alpha * hv.w);
}

// ---------------- one AGNN layer ----------------
static void run_layer(const float* xin, const float* W, const float* b, const float* beta,
                      const int* srcI, const int* dstI, int N, int E,
                      _Float16* A16, _Float16* Wp, float* h, float* hn,
                      float* sc, float* den, float* out, hipStream_t stream) {
  int nf = N * 128;
  k_pack_w   <<<64, 256, 0, stream>>>(W, Wp);
  k_cvt_f16  <<<(nf + 255) / 256, 256, 0, stream>>>(xin, A16, nf);
  k_gemm_wmma<<<(N + 15) / 16, 256, 0, stream>>>(A16, Wp, b, h, N);
  k_rownorm  <<<(N + 7) / 8, 256, 0, stream>>>(h, hn, N);
  k_zero     <<<(N + 255) / 256, 256, 0, stream>>>(den, N);
  k_zero     <<<(nf + 255) / 256, 256, 0, stream>>>(out, nf);
  k_edge_score<<<(E + 7) / 8, 256, 0, stream>>>(hn, srcI, dstI, beta, sc, den, E);
  k_edge_agg <<<(E + 7) / 8, 256, 0, stream>>>(h, sc, den, srcI, dstI, out, E);
  k_relu     <<<(nf + 255) / 256, 256, 0, stream>>>(out, nf);
}

extern "C" void kernel_launch(void* const* d_in, const int* in_sizes, int n_in,
                              void* d_out, int out_size, void* d_ws, size_t ws_size,
                              hipStream_t stream) {
  (void)n_in; (void)out_size; (void)ws_size;
  const float* x   = (const float*)d_in[0];
  const int*   ei  = (const int*)d_in[1];     // [2,E]: row0=src, row1=dst
  const float* W1  = (const float*)d_in[2];
  const float* b1  = (const float*)d_in[3];
  const float* be1 = (const float*)d_in[4];
  const float* W2  = (const float*)d_in[5];
  const float* b2  = (const float*)d_in[6];
  const float* be2 = (const float*)d_in[7];

  int N = in_sizes[0] / 128;
  int E = in_sizes[1] / 2;
  const int* srcI = ei;
  const int* dstI = ei + E;

  char* p = (char*)d_ws;
  auto carve = [&](size_t bytes) -> char* {
    char* q = p; p += (bytes + 255) & ~(size_t)255; return q;
  };
  _Float16* A16 = (_Float16*)carve((size_t)N * 128 * 2);
  _Float16* Wp  = (_Float16*)carve((size_t)16384 * 2);
  float* h   = (float*)carve((size_t)N * 128 * 4);
  float* hn  = (float*)carve((size_t)N * 128 * 4);
  float* y   = (float*)carve((size_t)N * 128 * 4);
  float* sc  = (float*)carve((size_t)E * 4);
  float* den = (float*)carve((size_t)N * 4);

  run_layer(x, W1, b1, be1, srcI, dstI, N, E, A16, Wp, h, hn, sc, den, y, stream);
  run_layer(y, W2, b2, be2, srcI, dstI, N, E, A16, Wp, h, hn, sc, den, (float*)d_out, stream);
}
